// MLLABlock_90512140796375
// MI455X (gfx1250) — matrix operand order repaired and checked by reading, not compiled, not run on hardware
//
#include <hip/hip_runtime.h>
#include <hip/hip_bf16.h>
#include <math.h>

// ---------------------------------------------------------------------------
// MLLA block for MI455X (gfx1250, wave32, WMMA).
// All dense GEMMs + linear-attention contractions run on
// v_wmma_f32_16x16x32_f16 with fp32 accumulation. WMMA operands (all but attn_av's q_rope A) are read
// from LDS in *fragment order* (2 x ds_load_b128 per lane); transposes are
// performed once at staging time with coalesced b128 global loads.
// ---------------------------------------------------------------------------

typedef __attribute__((ext_vector_type(16))) _Float16 v16h;
typedef __attribute__((ext_vector_type(8)))  _Float16 v8h;
typedef __attribute__((ext_vector_type(2)))  _Float16 v2h;
typedef __attribute__((ext_vector_type(8)))  float    v8f;

#define DEV static __device__ __forceinline__

DEV v8f wmma_16x16x32(v16h a, v16h b, v8f c) {
  return __builtin_amdgcn_wmma_f32_16x16x32_f16(
      /*neg_a=*/false, a, /*neg_b=*/false, b,
      /*c_mod=*/(short)0, c, /*reuse_a=*/false, /*reuse_b=*/false);
}

// Read a 16-half fragment stored contiguously (2 x b128 loads).
DEV v16h read_frag(const _Float16* p) {
  v8h lo = *(const v8h*)p;
  v8h hi = *(const v8h*)(p + 8);
  return __builtin_shufflevector(lo, hi, 0, 1, 2, 3, 4, 5, 6, 7, 8, 9, 10, 11,
                                 12, 13, 14, 15);
}

// Row-major f16 A fragment when stride%8==0: two contiguous 8-half runs.
// a[i] <-> k = (i<8 ? i : i+8) + hlf*8, so runs start at hlf*8 and 16+hlf*8.
DEV v16h load_frag_a_rm_vec(const _Float16* base, int stride) {
  const int lane = threadIdx.x & 31;
  const int hlf = lane >> 4, row = lane & 15;
  const _Float16* p = base + row * stride + hlf * 8;
  v8h lo = *(const v8h*)p;
  v8h hi = *(const v8h*)(p + 16);
  return __builtin_shufflevector(lo, hi, 0, 1, 2, 3, 4, 5, 6, 7, 8, 9, 10, 11,
                                 12, 13, 14, 15);
}

// Load 16 consecutive elements from global and convert to two v8h.
DEV void load16(const float* p, v8h& lo, v8h& hi) {
  const float4 f0 = *(const float4*)p;
  const float4 f1 = *(const float4*)(p + 4);
  const float4 f2 = *(const float4*)(p + 8);
  const float4 f3 = *(const float4*)(p + 12);
  lo[0] = (_Float16)f0.x; lo[1] = (_Float16)f0.y;
  lo[2] = (_Float16)f0.z; lo[3] = (_Float16)f0.w;
  lo[4] = (_Float16)f1.x; lo[5] = (_Float16)f1.y;
  lo[6] = (_Float16)f1.z; lo[7] = (_Float16)f1.w;
  hi[0] = (_Float16)f2.x; hi[1] = (_Float16)f2.y;
  hi[2] = (_Float16)f2.z; hi[3] = (_Float16)f2.w;
  hi[4] = (_Float16)f3.x; hi[5] = (_Float16)f3.y;
  hi[6] = (_Float16)f3.z; hi[7] = (_Float16)f3.w;
}
DEV void load16(const _Float16* p, v8h& lo, v8h& hi) {
  lo = *(const v8h*)p;
  hi = *(const v8h*)(p + 8);
}

enum { EPI_NONE = 0, EPI_SILU = 1, EPI_ELU1 = 2, EPI_GELU = 3 };

template <int EPI>
DEV float apply_epi(float v) {
  if (EPI == EPI_SILU) return v / (1.f + __expf(-v));
  if (EPI == EPI_ELU1) return v > 0.f ? v + 1.f : __expf(v);  // elu(v)+1
  if (EPI == EPI_GELU) return 0.5f * v * (1.f + erff(v * 0.70710678118f));
  return v;
}

// ---------------------------------------------------------------------------
// WMMA GEMM: out(MxN) = epi(A(MxK) @ B(KxN) + bias) [+ res]
// A: fp32/f16 row-major, converted + swizzled into fragment-order LDS.
// Bsw: weights pre-swizzled in global as [kt][nt][lane][16] f16.
// Tile 128x64, BK=32, 256 thr = 8 waves; wave w owns rows [16w,16w+16).
// LDS rows use a 24-half stride: 16B aligned for b128, bank-skewed.
// ---------------------------------------------------------------------------
template <typename AT, typename OT, int EPI, bool HAS_RES>
__global__ __launch_bounds__(256) void gemm_wmma(
    const AT* __restrict__ A, const _Float16* __restrict__ Bsw,
    const float* __restrict__ bias, const float* __restrict__ res,
    OT* __restrict__ out, int Mr, int Nn, int Kk) {
  __shared__ _Float16 As[8][32][24];  // 8 wave-tiles, fragment order
  __shared__ _Float16 Bs[4][32][24];  // 4 n-tiles, fragment order
  const int tid = threadIdx.x;
  const int wave = tid >> 5, lane = tid & 31;
  const long long bm = (long long)blockIdx.x * 128;
  const int bn = blockIdx.y * 64;
  const int ntiles = Nn >> 4;

  v8f acc[4] = {};

  // A staging: thread loads A[arow][k0+acb .. +16]; first 8 halves ->
  // LDS lane=row (frag K 0..7 / 16..23), next 8 -> lane=row+16.
  const int arow = tid >> 1, acb = (tid & 1) * 16;
  const int wt = arow >> 4, rr = arow & 15, ai = acb >> 1;  // ai in {0,8}
  // B staging: contiguous 2048-half copy from swizzled global
  const int bsft = tid * 8;
  const int btile = bsft >> 9, blane = (bsft >> 4) & 31, bi = bsft & 15;

  for (int k0 = 0; k0 < Kk; k0 += 32) {
    v8h alo, ahi;
    load16(A + (bm + arow) * (long long)Kk + k0 + acb, alo, ahi);
    *(v8h*)&As[wt][rr][ai] = alo;
    *(v8h*)&As[wt][rr + 16][ai] = ahi;
    const _Float16* bsrc =
        Bsw + (((long long)(k0 >> 5) * ntiles + (bn >> 4)) << 9) + bsft;
    *(v8h*)&Bs[btile][blane][bi] = *(const v8h*)bsrc;
    if (k0 + 32 < Kk)  // global_prefetch_b8 of next A tile
      __builtin_prefetch(A + (bm + arow) * (long long)Kk + k0 + 32 + acb, 0, 3);
    __syncthreads();

    v16h af = read_frag(&As[wave][lane][0]);
#pragma unroll
    for (int t = 0; t < 4; ++t) {
      v16h bf = read_frag(&Bs[t][lane][0]);
      acc[t] = wmma_16x16x32(af, bf, acc[t]);
    }
    __syncthreads();
  }

  const int hlf = lane >> 4, col = lane & 15;
#pragma unroll
  for (int t = 0; t < 4; ++t) {
#pragma unroll
    for (int g = 0; g < 8; ++g) {
      const long long row = bm + wave * 16 + g + hlf * 8;
      const int c = bn + t * 16 + col;
      float v = acc[t][g] + bias[c];
      v = apply_epi<EPI>(v);
      if (HAS_RES) v += res[row * Nn + c];
      out[row * Nn + c] = (OT)v;
    }
  }
}

// fp32 -> f16 weight conversion into the swizzled [kt][nt][lane][16] layout.
__global__ __launch_bounds__(256) void cvt_swz_b_kernel(
    const float* __restrict__ in, _Float16* __restrict__ out, int K, int N) {
  const long long total = (long long)K * N;
  const int ntiles = N >> 4;
  for (long long idx = (long long)blockIdx.x * 256 + threadIdx.x; idx < total;
       idx += (long long)gridDim.x * 256) {
    const int n = (int)(idx % N);
    const int k = (int)(idx / N);
    const int kt = k >> 5, nt = n >> 4;
    const int lane = ((k >> 3) & 1) * 16 + (n & 15);
    const int i = (k & 7) + 8 * ((k >> 4) & 1);
    out[((((long long)kt * ntiles + nt) * 32 + lane) << 4) + i] =
        (_Float16)in[idx];
  }
}

// ---------------------------------------------------------------------------
// Depthwise 3x3 conv (channel-last, SAME), float4 vectorized.
// MODE 0: out = in + (conv+bias); MODE 1: silu(conv+bias); MODE 2: conv+bias
// ---------------------------------------------------------------------------
template <int MODE>
__global__ __launch_bounds__(256) void dwconv3x3_kernel(
    const float* __restrict__ in, const float* __restrict__ wgt,
    const float* __restrict__ bias, float* __restrict__ out,
    int Bn, int Hh, int Ww, int Cc) {
  const int c4n = Cc >> 2;
  const long long total = (long long)Bn * Hh * Ww * c4n;
  long long idx = (long long)blockIdx.x * blockDim.x + threadIdx.x;
  if (idx >= total) return;
  const int c4 = (int)(idx % c4n);
  long long pix = idx / c4n;
  const int xw = (int)(pix % Ww);
  pix /= Ww;
  const int yh = (int)(pix % Hh);
  const int b = (int)(pix / Hh);
  const int c0 = c4 * 4;
  float4 acc = make_float4(bias[c0], bias[c0 + 1], bias[c0 + 2], bias[c0 + 3]);
#pragma unroll
  for (int dy = -1; dy <= 1; ++dy) {
#pragma unroll
    for (int dx = -1; dx <= 1; ++dx) {
      const int yy = yh + dy, xx = xw + dx;
      if (yy < 0 || yy >= Hh || xx < 0 || xx >= Ww) continue;
      const float4 p =
          *(const float4*)(in + (((long long)b * Hh + yy) * Ww + xx) * Cc + c0);
      const int tap = (dy + 1) * 3 + (dx + 1);
      acc.x += p.x * wgt[(c0 + 0) * 9 + tap];
      acc.y += p.y * wgt[(c0 + 1) * 9 + tap];
      acc.z += p.z * wgt[(c0 + 2) * 9 + tap];
      acc.w += p.w * wgt[(c0 + 3) * 9 + tap];
    }
  }
  const long long o = (((long long)b * Hh + yh) * Ww + xw) * Cc + c0;
  float4 r;
  if (MODE == 0) {
    const float4 cen = *(const float4*)(in + o);
    r = make_float4(cen.x + acc.x, cen.y + acc.y, cen.z + acc.z, cen.w + acc.w);
  } else if (MODE == 1) {
    r = make_float4(acc.x / (1.f + __expf(-acc.x)),
                    acc.y / (1.f + __expf(-acc.y)),
                    acc.z / (1.f + __expf(-acc.z)),
                    acc.w / (1.f + __expf(-acc.w)));
  } else {
    r = acc;
  }
  *(float4*)(out + o) = r;
}

// LayerNorm over C=192, one wave per token (6 channels/lane), f16 output.
__global__ __launch_bounds__(256) void layernorm_kernel(
    const float* __restrict__ x, const float* __restrict__ g,
    const float* __restrict__ be, _Float16* __restrict__ out, long long Mr) {
  const int wave = threadIdx.x >> 5, lane = threadIdx.x & 31;
  const long long row = (long long)blockIdx.x * 8 + wave;
  if (row >= Mr) return;
  const float* xr = x + row * 192;
  float v[6];
  float s = 0.f;
#pragma unroll
  for (int i = 0; i < 6; ++i) { v[i] = xr[lane + i * 32]; s += v[i]; }
#pragma unroll
  for (int off = 16; off > 0; off >>= 1) s += __shfl_xor(s, off, 32);
  const float mu = s * (1.f / 192.f);
  float s2 = 0.f;
#pragma unroll
  for (int i = 0; i < 6; ++i) { const float d = v[i] - mu; s2 += d * d; }
#pragma unroll
  for (int off = 16; off > 0; off >>= 1) s2 += __shfl_xor(s2, off, 32);
  const float inv = rsqrtf(s2 * (1.f / 192.f) + 1e-5f);
#pragma unroll
  for (int i = 0; i < 6; ++i) {
    const int c = lane + i * 32;
    out[row * 192 + c] = (_Float16)((v[i] - mu) * inv * g[c] + be[c]);
  }
}

// kmean[b][c] = mean_n k[b,n,c]; k is the elu+1 half of qk (B,N,2C)
__global__ __launch_bounds__(256) void kmean_kernel(
    const float* __restrict__ qk, float* __restrict__ km, int Ntok, int Cc) {
  const int b = blockIdx.x / Cc, c = blockIdx.x % Cc;
  float s = 0.f;
  for (int n = threadIdx.x; n < Ntok; n += 256)
    s += qk[((long long)b * Ntok + n) * (2 * Cc) + Cc + c];
  __shared__ float red[256];
  red[threadIdx.x] = s;
  __syncthreads();
  for (int off = 128; off > 0; off >>= 1) {
    if (threadIdx.x < off) red[threadIdx.x] += red[threadIdx.x + off];
    __syncthreads();
  }
  if (threadIdx.x == 0) km[b * Cc + c] = red[0] / (float)Ntok;
}

// 2D axial RoPE applied to q and k; paired loads/stores; f16 outputs.
__global__ __launch_bounds__(256) void rope_kernel(
    const float* __restrict__ qk, _Float16* __restrict__ qr,
    _Float16* __restrict__ kr, int Bn, int Hh, int Ww, int Cc) {
  const int halfc = Cc >> 1, kmax = Cc >> 2;
  const long long total = (long long)Bn * Hh * Ww * halfc;
  for (long long idx = (long long)blockIdx.x * 256 + threadIdx.x; idx < total;
       idx += (long long)gridDim.x * 256) {
    const int j = (int)(idx % halfc);
    const long long nn = idx / halfc;
    const int n = (int)(nn % (Hh * Ww));
    const int b = (int)(nn / (Hh * Ww));
    const int y = n / Ww, xw2 = n % Ww;
    const int jj = (j < kmax) ? j : j - kmax;
    const float theta = __powf(10000.f, -(float)jj / (float)kmax);
    const float pos = (j < kmax) ? (float)y : (float)xw2;
    float sn, cs;
    __sincosf(pos * theta, &sn, &cs);
    const long long base = (long long)b * (Hh * Ww) + n;
    const float* row = qk + base * (2 * Cc);
    const float2 qv = *(const float2*)(row + 2 * j);
    const float2 kvp = *(const float2*)(row + Cc + 2 * j);
    const long long ob = base * Cc + 2 * j;
    v2h qo, ko;
    qo[0] = (_Float16)(qv.x * cs - qv.y * sn);
    qo[1] = (_Float16)(qv.y * cs + qv.x * sn);
    ko[0] = (_Float16)(kvp.x * cs - kvp.y * sn);
    ko[1] = (_Float16)(kvp.y * cs + kvp.x * sn);
    *(v2h*)(qr + ob) = qo;
    *(v2h*)(kr + ob) = ko;
  }
}

__global__ __launch_bounds__(256) void zero_f32(float* p, long long n) {
  for (long long i = (long long)blockIdx.x * 256 + threadIdx.x; i < n;
       i += (long long)gridDim.x * 256)
    p[i] = 0.f;
}

// Vectorized fp32 -> f16 (n must be a multiple of 8).
__global__ __launch_bounds__(256) void cvt_f32_f16_v8(
    const float* __restrict__ in, _Float16* __restrict__ out, long long n8) {
  for (long long i = (long long)blockIdx.x * 256 + threadIdx.x; i < n8;
       i += (long long)gridDim.x * 256) {
    const float4 a = ((const float4*)in)[2 * i];
    const float4 b = ((const float4*)in)[2 * i + 1];
    v8h o;
    o[0] = (_Float16)a.x; o[1] = (_Float16)a.y;
    o[2] = (_Float16)a.z; o[3] = (_Float16)a.w;
    o[4] = (_Float16)b.x; o[5] = (_Float16)b.y;
    o[6] = (_Float16)b.z; o[7] = (_Float16)b.w;
    ((v8h*)out)[i] = o;
  }
}

// ---------------------------------------------------------------------------
// kv[b,h,d,e] = (1/N) sum_n k_rope[b,n,h,d] * v[b,n,h,e]  via WMMA.
// grid (B*NH, 64). Each of 256 threads owns one token row: 4+4 coalesced
// b128 global loads, then scatters halves into fragment-order LDS tiles
// (the transpose happens here, once). Each wave then reads its operands as
// 2 x ds_load_b128 and runs 4 WMMAs; ds_add_f32 block reduce + global atomics.
// ---------------------------------------------------------------------------
__global__ __launch_bounds__(256) void kv_kernel(
    const _Float16* __restrict__ kr, const _Float16* __restrict__ v16,
    float* __restrict__ kv, int NH, int Ntok, int Cc) {
  const int bh = blockIdx.x;
  const int b = bh / NH, h = bh % NH;
  __shared__ _Float16 Ks[8][2][32][24];  // [wave][mtile][lane][frag]
  __shared__ _Float16 Vs[8][2][32][24];  // [wave][etile][lane][frag]
  __shared__ float kred[1024];
  for (int i = threadIdx.x; i < 1024; i += 256) kred[i] = 0.f;

  const int w = threadIdx.x >> 5, kk = threadIdx.x & 31;
  const long long n = (long long)blockIdx.y * 256 + threadIdx.x;
  const _Float16* krow = kr + ((long long)b * Ntok + n) * Cc + h * 32;
  const _Float16* vrow = v16 + ((long long)b * Ntok + n) * Cc + h * 32;
  v8h kvv[4], vvv[4];
#pragma unroll
  for (int j = 0; j < 4; ++j) {
    kvv[j] = *(const v8h*)(krow + 8 * j);
    vvv[j] = *(const v8h*)(vrow + 8 * j);
  }
  // A-side (k^T): element (m=d, k=kk) -> lane = (d&15)+16*((kk>>3)&1),
  //               frag idx = (kk&7)+8*(kk>>4)
  // B-side (v):   element (k=kk, n=d) -> lane = (d&15)+16*((kk>>3)&1),
  //               frag idx = (kk&7)+8*(kk>>4)
  const int la = 16 * ((kk >> 3) & 1), ia = (kk & 7) + 8 * (kk >> 4);
  const int lb = 16 * ((kk >> 3) & 1), ib = (kk & 7) + 8 * (kk >> 4);
#pragma unroll
  for (int j = 0; j < 4; ++j)
#pragma unroll
    for (int u = 0; u < 8; ++u) {
      const int d = j * 8 + u;
      Ks[w][d >> 4][(d & 15) + la][ia] = kvv[j][u];
      Vs[w][d >> 4][(d & 15) + lb][ib] = vvv[j][u];
    }
  __syncthreads();

  const int lane = threadIdx.x & 31;
  const int hlf = lane >> 4, r = lane & 15;
  v16h a0 = read_frag(&Ks[w][0][lane][0]);
  v16h a1 = read_frag(&Ks[w][1][lane][0]);
  v16h b0 = read_frag(&Vs[w][0][lane][0]);
  v16h b1 = read_frag(&Vs[w][1][lane][0]);

  v8f acc[2][2] = {};
  acc[0][0] = wmma_16x16x32(a0, b0, acc[0][0]);
  acc[0][1] = wmma_16x16x32(a0, b1, acc[0][1]);
  acc[1][0] = wmma_16x16x32(a1, b0, acc[1][0]);
  acc[1][1] = wmma_16x16x32(a1, b1, acc[1][1]);

#pragma unroll
  for (int mi = 0; mi < 2; ++mi)
#pragma unroll
    for (int ei = 0; ei < 2; ++ei)
#pragma unroll
      for (int g = 0; g < 8; ++g)
        atomicAdd(&kred[(mi * 16 + g + hlf * 8) * 32 + ei * 16 + r],
                  acc[mi][ei][g]);
  __syncthreads();

  const float invn = 1.f / (float)Ntok;  // folds the s*s = 1/N scaling
  float* kvp = kv + (long long)bh * 1024;
  for (int i = threadIdx.x; i < 1024; i += 256)
    atomicAdd(&kvp[i], kred[i] * invn);
}

// ---------------------------------------------------------------------------
// a = (q_rope @ kv) * z + lepe ; t = a * act_res  (per head, WMMA)
// kv staged into fragment-order LDS; q_rope fragments are two contiguous
// 16B global loads per lane. z spread by lane shuffle.
// ---------------------------------------------------------------------------
__global__ __launch_bounds__(256) void attn_av_kernel(
    const _Float16* __restrict__ qr, const float* __restrict__ qk,
    const float* __restrict__ kv, const float* __restrict__ km,
    const float* __restrict__ lepe, const _Float16* __restrict__ act16,
    float* __restrict__ tout, int NH, int Ntok, int Cc) {
  const int bh = blockIdx.x;
  const int b = bh / NH, h = bh % NH;
  __shared__ _Float16 kvs[2][32][24];  // [etile][lane][frag], fragment order
  for (int i = threadIdx.x; i < 1024; i += 256) {
    const int d = i >> 5, e = i & 31;
    kvs[e >> 4][(e & 15) + 16 * ((d >> 3) & 1)][(d & 7) + 8 * (d >> 4)] =
        (_Float16)kv[(long long)bh * 1024 + i];
  }
  __syncthreads();

  const int wave = threadIdx.x >> 5, lane = threadIdx.x & 31;
  const int hlf = lane >> 4, r = lane & 15;
  const int n0 = blockIdx.y * 128 + wave * 16;

  float zval;
  {
    const int n = n0 + r;
    const float* qp = qk + ((long long)b * Ntok + n) * (2 * Cc) + h * 32;
    const float* kmp = km + b * Cc + h * 32;
    float s = 0.f;
#pragma unroll
    for (int d = 0; d < 32; ++d) s += qp[d] * kmp[d];
    zval = 1.f / (s + 1e-6f);
  }

  const _Float16* qb = qr + ((long long)b * Ntok + n0) * Cc + h * 32;
  v16h afrag = load_frag_a_rm_vec(qb, Cc);
  v8f acc[2] = {};
  acc[0] = wmma_16x16x32(afrag, read_frag(&kvs[0][lane][0]), acc[0]);
  acc[1] = wmma_16x16x32(afrag, read_frag(&kvs[1][lane][0]), acc[1]);

#pragma unroll
  for (int t = 0; t < 2; ++t) {
#pragma unroll
    for (int g = 0; g < 8; ++g) {
      const int m = g + hlf * 8;
      const float z = __shfl(zval, m, 32);
      const long long o =
          ((long long)b * Ntok + n0 + m) * Cc + h * 32 + t * 16 + r;
      const float aval = acc[t][g] * z + lepe[o];
      tout[o] = aval * (float)act16[o];
    }
  }
}

// ---------------------------------------------------------------------------
// Host orchestration
// ---------------------------------------------------------------------------
extern "C" void kernel_launch(void* const* d_in, const int* in_sizes, int n_in,
                              void* d_out, int out_size, void* d_ws,
                              size_t ws_size, hipStream_t stream) {
  (void)in_sizes; (void)n_in; (void)out_size; (void)ws_size;
  const int B = 8, H = 128, W = 128, C = 192, NH = 6;
  const long long N = (long long)H * W;   // 16384
  const long long M = (long long)B * N;   // 131072 tokens

  const float* x       = (const float*)d_in[0];
  const float* cpe1_w  = (const float*)d_in[1];
  const float* cpe1_b  = (const float*)d_in[2];
  const float* norm1_g = (const float*)d_in[3];
  const float* norm1_b = (const float*)d_in[4];
  const float* act_w   = (const float*)d_in[5];
  const float* act_b   = (const float*)d_in[6];
  const float* in_w    = (const float*)d_in[7];
  const float* in_b    = (const float*)d_in[8];
  const float* dwc_w   = (const float*)d_in[9];
  const float* dwc_b   = (const float*)d_in[10];
  const float* qk_w    = (const float*)d_in[11];
  const float* qk_b    = (const float*)d_in[12];
  const float* lepe_w  = (const float*)d_in[13];
  const float* lepe_b  = (const float*)d_in[14];
  const float* out_w   = (const float*)d_in[15];
  const float* out_b   = (const float*)d_in[16];
  const float* cpe2_w  = (const float*)d_in[17];
  const float* cpe2_b  = (const float*)d_in[18];
  const float* norm2_g = (const float*)d_in[19];
  const float* norm2_b = (const float*)d_in[20];
  const float* fc1_w   = (const float*)d_in[21];
  const float* fc1_b   = (const float*)d_in[22];
  const float* fc2_w   = (const float*)d_in[23];
  const float* fc2_b   = (const float*)d_in[24];

  // ---- workspace arena (aliased; ~704 MB) ----
  const size_t F32 = (size_t)M * C * sizeof(float);      // 96 MiB
  const size_t F16 = (size_t)M * C * sizeof(_Float16);   // 48 MiB
  uint8_t* w = (uint8_t*)d_ws;
  float*    x1    = (float*)(w);                          // shortcut
  _Float16* xn16  = (_Float16*)(w + F32);                 // xn -> q_rope -> xn2
  _Float16* act16 = (_Float16*)(w + F32 + F16);           // silu(act) f16
  float*    xi0   = (float*)(w + F32 + 2 * F16);          // xi0 -> lepe -> x3
  float*    vbuf  = (float*)(w + 2 * F32 + 2 * F16);      // v -> t
  float*    qkbuf = (float*)(w + 3 * F32 + 2 * F16);      // qk(2C) -> x2 -> h1
  _Float16* v16   = (_Float16*)(w + 5 * F32 + 2 * F16);
  _Float16* kr16  = (_Float16*)(w + 5 * F32 + 3 * F16);
  float*    kvbuf = (float*)(w + 5 * F32 + 4 * F16);      // 48*1024 f32
  float*    kmeanb = kvbuf + (size_t)B * NH * 1024;       // B*C f32
  _Float16* wts   = (_Float16*)(kmeanb + (size_t)B * C);
  _Float16* act_w16 = wts;
  _Float16* in_w16  = act_w16 + (size_t)C * C;
  _Float16* qk_w16  = in_w16 + (size_t)C * C;
  _Float16* out_w16 = qk_w16 + (size_t)C * 2 * C;
  _Float16* fc1_w16 = out_w16 + (size_t)C * C;
  _Float16* fc2_w16 = fc1_w16 + (size_t)C * 4 * C;

  // aliases (lifetimes verified: each slot is dead before reuse)
  _Float16* qr16 = xn16;
  float*    lepeb = xi0;
  float*    tbuf = vbuf;
  float*    x2   = qkbuf;
  float*    x3   = xi0;
  _Float16* xn2  = xn16;
  _Float16* h1   = (_Float16*)qkbuf;

  const dim3 blk(256);
  const int dwBlocks = (int)((M * (C / 4) + 255) / 256);  // 24576
  const int lnBlocks = (int)(M / 8);                      // 16384

  // 0) weight conversion fp32 -> f16 in WMMA-fragment-swizzled layout
  cvt_swz_b_kernel<<<64, blk, 0, stream>>>(act_w, act_w16, C, C);
  cvt_swz_b_kernel<<<64, blk, 0, stream>>>(in_w, in_w16, C, C);
  cvt_swz_b_kernel<<<128, blk, 0, stream>>>(qk_w, qk_w16, C, 2 * C);
  cvt_swz_b_kernel<<<64, blk, 0, stream>>>(out_w, out_w16, C, C);
  cvt_swz_b_kernel<<<256, blk, 0, stream>>>(fc1_w, fc1_w16, C, 4 * C);
  cvt_swz_b_kernel<<<256, blk, 0, stream>>>(fc2_w, fc2_w16, 4 * C, C);

  // 1) cpe1 residual: x1 = x + dwconv(x)
  dwconv3x3_kernel<0><<<dwBlocks, blk, 0, stream>>>(x, cpe1_w, cpe1_b, x1, B, H, W, C);
  // 2) LN1 -> f16
  layernorm_kernel<<<lnBlocks, blk, 0, stream>>>(x1, norm1_g, norm1_b, xn16, M);
  // 3) act_res = silu(xn @ act_w + b) (f16 out)
  gemm_wmma<_Float16, _Float16, EPI_SILU, false><<<dim3((int)(M / 128), C / 64), blk, 0, stream>>>(
      xn16, act_w16, act_b, nullptr, act16, (int)M, C, C);
  // 4) xi0 = xn @ in_w + b
  gemm_wmma<_Float16, float, EPI_NONE, false><<<dim3((int)(M / 128), C / 64), blk, 0, stream>>>(
      xn16, in_w16, in_b, nullptr, xi0, (int)M, C, C);
  // 5) v = silu(dwconv(xi0))
  dwconv3x3_kernel<1><<<dwBlocks, blk, 0, stream>>>(xi0, dwc_w, dwc_b, vbuf, B, H, W, C);
  cvt_f32_f16_v8<<<4096, blk, 0, stream>>>(vbuf, v16, M * C / 8);
  // 6) qk = elu(v @ qk_w + b)+1  (B,N,2C)
  gemm_wmma<float, float, EPI_ELU1, false><<<dim3((int)(M / 128), 2 * C / 64), blk, 0, stream>>>(
      vbuf, qk_w16, qk_b, nullptr, qkbuf, (int)M, 2 * C, C);
  // 7) kmean over N
  kmean_kernel<<<B * C, blk, 0, stream>>>(qkbuf, kmeanb, (int)N, C);
  // 8) RoPE -> q_rope16, k_rope16
  rope_kernel<<<8192, blk, 0, stream>>>(qkbuf, qr16, kr16, B, H, W, C);
  // 9) kv accumulation (WMMA + ds_add_f32 + global atomics)
  zero_f32<<<192, blk, 0, stream>>>(kvbuf, (long long)B * NH * 1024);
  kv_kernel<<<dim3(B * NH, (int)(N / 256)), blk, 0, stream>>>(kr16, v16, kvbuf, NH, (int)N, C);
  // 10) lepe = dwconv(v) + bias
  dwconv3x3_kernel<2><<<dwBlocks, blk, 0, stream>>>(vbuf, lepe_w, lepe_b, lepeb, B, H, W, C);
  // 11) t = ((q_rope@kv)*z + lepe) * act_res  (overwrites v slot)
  attn_av_kernel<<<dim3(B * NH, (int)(N / 128)), blk, 0, stream>>>(
      qr16, qkbuf, kvbuf, kmeanb, lepeb, act16, tbuf, NH, (int)N, C);
  // 12) x2 = shortcut + t @ out_w + b
  gemm_wmma<float, float, EPI_NONE, true><<<dim3((int)(M / 128), C / 64), blk, 0, stream>>>(
      tbuf, out_w16, out_b, x1, x2, (int)M, C, C);
  // 13) cpe2 residual: x3 = x2 + dwconv(x2)
  dwconv3x3_kernel<0><<<dwBlocks, blk, 0, stream>>>(x2, cpe2_w, cpe2_b, x3, B, H, W, C);
  // 14) LN2 -> f16
  layernorm_kernel<<<lnBlocks, blk, 0, stream>>>(x3, norm2_g, norm2_b, xn2, M);
  // 15) h1 = gelu(xn2 @ fc1_w + b) (f16, reuses qk slot)
  gemm_wmma<_Float16, _Float16, EPI_GELU, false><<<dim3((int)(M / 128), 4 * C / 64), blk, 0, stream>>>(
      xn2, fc1_w16, fc1_b, nullptr, h1, (int)M, 4 * C, C);
  // 16) out = x3 + h1 @ fc2_w + b
  gemm_wmma<_Float16, float, EPI_NONE, true><<<dim3((int)(M / 128), C / 64), blk, 0, stream>>>(
      h1, fc2_w16, fc2_b, x3, (float*)d_out, (int)M, C, 4 * C);
}
